// SRNN_Softmax_7421703487923
// MI455X (gfx1250) — matrix operand order repaired and checked by reading, not compiled, not run on hardware
//
#include <hip/hip_runtime.h>
#include <math.h>

// Problem dims (match reference)
#define HID_N   1024
#define OUT_N   512
#define VOCAB_N 1024
#define MSIZE_N 262144
#define MDIM_N  128

typedef __attribute__((ext_vector_type(16))) _Float16 v16h;
typedef __attribute__((ext_vector_type(8)))  float    v8f;
typedef __attribute__((ext_vector_type(4)))  float    fvec4;

// ---------------------------------------------------------------------------
// WMMA GEMV helpers.
// y[o] = sum_k x[k] * W[o,k]   computed as D = A(16x32) x B(32x16) + C with
// A row M=0 = x[k0..k0+31] (other rows zero), B[kk,n] = W[o0+n, k0+kk].
// A layout used (16x32, matches B's contiguous K order): lanes 0..15 hold M=lane with
// K = 0..15; lanes 16..31 hold M=lane-16 with K = 16..31.
// Only lanes 0 and 16 carry row M=0 data; all other lanes supply zeros.
// ---------------------------------------------------------------------------
__device__ __forceinline__ v16h make_A_rowvec(const float* __restrict__ x, int k0) {
  const int lane = threadIdx.x & 31;
  v16h a = {};
  if (lane == 0) {
#pragma unroll
    for (int e = 0; e < 16; ++e) {
      const int k = e;                              // K = 0..15
      a[e] = (_Float16)x[k0 + k];
    }
  } else if (lane == 16) {
#pragma unroll
    for (int e = 0; e < 16; ++e) {
      const int k = e + 16;                         // K = 16..31
      a[e] = (_Float16)x[k0 + k];
    }
  }
  return a;
}

// B (32x16, K x N): lane n = column, lanes 0..15 hold K=0..15, lanes 16..31
// hold K=16..31. Contiguous 64B row read from W (row-major, ld = K).
__device__ __forceinline__ v16h make_B_tile(const float* __restrict__ W, int ld,
                                            int o0, int k0) {
  const int lane = threadIdx.x & 31;
  const int n  = lane & 15;
  const int kb = k0 + ((lane & 16) ? 16 : 0);
  const fvec4* p4 = reinterpret_cast<const fvec4*>(W + (size_t)(o0 + n) * ld + kb);
  const fvec4 q0 = p4[0], q1 = p4[1], q2 = p4[2], q3 = p4[3];
  v16h b;
  b[0]  = (_Float16)q0.x; b[1]  = (_Float16)q0.y; b[2]  = (_Float16)q0.z; b[3]  = (_Float16)q0.w;
  b[4]  = (_Float16)q1.x; b[5]  = (_Float16)q1.y; b[6]  = (_Float16)q1.z; b[7]  = (_Float16)q1.w;
  b[8]  = (_Float16)q2.x; b[9]  = (_Float16)q2.y; b[10] = (_Float16)q2.z; b[11] = (_Float16)q2.w;
  b[12] = (_Float16)q3.x; b[13] = (_Float16)q3.y; b[14] = (_Float16)q3.z; b[15] = (_Float16)q3.w;
  return b;
}

// Accumulate y(o0..o0+15) += x . W[o0+n,:] over K via v_wmma_f32_16x16x32_f16
__device__ __forceinline__ v8f gemv_acc(v8f acc, const float* __restrict__ x,
                                        const float* __restrict__ W, int K, int o0) {
  for (int k0 = 0; k0 < K; k0 += 32) {
    v16h a = make_A_rowvec(x, k0);
    v16h b = make_B_tile(W, K, o0, k0);
    acc = __builtin_amdgcn_wmma_f32_16x16x32_f16(
        /*neg_a=*/false, a, /*neg_b=*/false, b,
        /*c_mod=*/(short)0, acc, /*reuse_a=*/false, /*reuse_b=*/false);
  }
  return acc;
}

__device__ __forceinline__ float sigmoidf(float v) { return 1.0f / (1.0f + expf(-v)); }

// ---------------------------------------------------------------------------
// Kernel 1: hidden_bar = stack[0] . W_sh^T + b_sh + hidden0     (1024 outs)
// ---------------------------------------------------------------------------
__global__ void k_hbar(const float* __restrict__ stack, const float* __restrict__ W_sh,
                       const float* __restrict__ b_sh, const float* __restrict__ hidden0,
                       float* __restrict__ hbar) {
  const int wave = blockIdx.x * (blockDim.x >> 5) + (threadIdx.x >> 5);
  const int o0 = wave * 16;
  if (o0 >= HID_N) return;
  v8f acc = {};
  acc = gemv_acc(acc, stack /* row 0 */, W_sh, MDIM_N, o0);
  const int lane = threadIdx.x & 31;
  if (lane < 16) {                       // D row M=0 lives in VGPR0, lanes 0..15
    const int o = o0 + lane;
    hbar[o] = acc[0] + b_sh[o] + hidden0[o];
  }
}

// ---------------------------------------------------------------------------
// Kernel 2: h = tanh(x . W_ih^T + b_ih + hbar . W_hh^T + b_hh)  (1024 outs)
// ---------------------------------------------------------------------------
__global__ void k_hidden(const float* __restrict__ x, const float* __restrict__ hbar,
                         const float* __restrict__ W_ih, const float* __restrict__ W_hh,
                         const float* __restrict__ b_ih, const float* __restrict__ b_hh,
                         float* __restrict__ h) {
  const int wave = blockIdx.x * (blockDim.x >> 5) + (threadIdx.x >> 5);
  const int o0 = wave * 16;
  if (o0 >= HID_N) return;
  v8f acc = {};
  acc = gemv_acc(acc, x,    W_ih, VOCAB_N, o0);
  acc = gemv_acc(acc, hbar, W_hh, HID_N,   o0);
  const int lane = threadIdx.x & 31;
  if (lane < 16) {
    const int o = o0 + lane;
    h[o] = tanhf(acc[0] + b_ih[o] + b_hh[o]);
  }
}

// ---------------------------------------------------------------------------
// Kernel 3: sigmoid head  y = sigma(h . W^T + b)   (used for W_y and W_n)
// ---------------------------------------------------------------------------
__global__ void k_head_sigmoid(const float* __restrict__ h, const float* __restrict__ W,
                               const float* __restrict__ b, float* __restrict__ y,
                               int nout, int K) {
  const int wave = blockIdx.x * (blockDim.x >> 5) + (threadIdx.x >> 5);
  const int o0 = wave * 16;
  if (o0 >= nout) return;
  v8f acc = {};
  acc = gemv_acc(acc, h, W, K, o0);
  const int lane = threadIdx.x & 31;
  if (lane < 16) {
    const int o = o0 + lane;
    y[o] = sigmoidf(acc[0] + b[o]);
  }
}

// ---------------------------------------------------------------------------
// Kernel 4: action weights = softmax(h . W_a^T + b_a), 2 logits.
// One wave per logit (wave32 shuffle reduction), then softmax on thread 0.
// ---------------------------------------------------------------------------
__global__ void k_action(const float* __restrict__ h, const float* __restrict__ W_a,
                         const float* __restrict__ b_a, float* __restrict__ aw) {
  __shared__ float logits[2];
  const int wave = threadIdx.x >> 5;
  const int lane = threadIdx.x & 31;
  if (wave < 2) {
    float s = 0.0f;
    for (int k = lane; k < HID_N; k += 32) s += h[k] * W_a[wave * HID_N + k];
#pragma unroll
    for (int m = 16; m > 0; m >>= 1) s += __shfl_xor(s, m, 32);
    if (lane == 0) logits[wave] = s + b_a[wave];
  }
  __syncthreads();
  if (threadIdx.x == 0) {
    const float l0 = logits[0], l1 = logits[1];
    const float mx = fmaxf(l0, l1);
    const float e0 = expf(l0 - mx), e1 = expf(l1 - mx);
    const float inv = 1.0f / (e0 + e1);
    aw[0] = e0 * inv;
    aw[1] = e1 * inv;
  }
}

// ---------------------------------------------------------------------------
// Kernel 5: stack blend — the bandwidth-dominant part (~268 MB traffic).
// new_stack[i] = a0*push[i] + a1*pop[i], one ext-vector float4 per thread,
// NT stores (output never re-read; stack reads get push/pop reuse in L2).
// ---------------------------------------------------------------------------
__global__ void k_stack(const float* __restrict__ stack, const float* __restrict__ new_elt,
                        const float* __restrict__ aw, float* __restrict__ out) {
  const float a0 = aw[0];
  const float a1 = aw[1];
  const size_t idx = (size_t)blockIdx.x * blockDim.x + threadIdx.x;   // float4 index
  const size_t row = idx >> 5;                                        // 32 float4 / row
  const int    c4  = (int)(idx & 31);
  const fvec4* s4 = reinterpret_cast<const fvec4*>(stack);

  fvec4 push, pop = {};
  if (row == 0) push = reinterpret_cast<const fvec4*>(new_elt)[c4];
  else          push = s4[(row - 1) * 32 + c4];
  if (row != (size_t)MSIZE_N - 1) pop = s4[(row + 1) * 32 + c4];

  const fvec4 r = a0 * push + a1 * pop;
  __builtin_nontemporal_store(r, reinterpret_cast<fvec4*>(out) + idx);
}

// ---------------------------------------------------------------------------
extern "C" void kernel_launch(void* const* d_in, const int* in_sizes, int n_in,
                              void* d_out, int out_size, void* d_ws, size_t ws_size,
                              hipStream_t stream) {
  (void)in_sizes; (void)n_in; (void)out_size; (void)ws_size;

  const float* input   = (const float*)d_in[0];
  const float* hidden0 = (const float*)d_in[1];
  const float* stack   = (const float*)d_in[2];
  const float* W_ih    = (const float*)d_in[3];
  const float* W_hh    = (const float*)d_in[4];
  const float* b_ih    = (const float*)d_in[5];
  const float* b_hh    = (const float*)d_in[6];
  const float* W_y     = (const float*)d_in[7];
  const float* b_y     = (const float*)d_in[8];
  const float* W_n     = (const float*)d_in[9];
  const float* b_n     = (const float*)d_in[10];
  const float* W_a     = (const float*)d_in[11];
  const float* b_a     = (const float*)d_in[12];
  const float* W_sh    = (const float*)d_in[13];
  const float* b_sh    = (const float*)d_in[14];

  float* out       = (float*)d_out;
  float* out_y     = out;                   // 512
  float* out_h     = out + OUT_N;           // 1024
  float* out_stack = out + OUT_N + HID_N;   // 262144*128 (16B-aligned offset)

  float* ws      = (float*)d_ws;
  float* hbar    = ws;                      // 1024 floats
  float* new_elt = ws + HID_N;              // 128 floats (16B-aligned)
  float* aw      = ws + HID_N + MDIM_N;     // 2 floats

  // 8 waves / block, 16 outputs / wave
  k_hbar        <<<HID_N / 128, 256, 0, stream>>>(stack, W_sh, b_sh, hidden0, hbar);
  k_hidden      <<<HID_N / 128, 256, 0, stream>>>(input, hbar, W_ih, W_hh, b_ih, b_hh, out_h);
  k_head_sigmoid<<<OUT_N / 128, 256, 0, stream>>>(out_h, W_y, b_y, out_y, OUT_N, HID_N);
  k_head_sigmoid<<<1,           256, 0, stream>>>(out_h, W_n, b_n, new_elt, MDIM_N, HID_N);
  k_action      <<<1,            64, 0, stream>>>(out_h, W_a, b_a, aw);

  const size_t n4 = (size_t)MSIZE_N * MDIM_N / 4;   // 8,388,608 float4
  k_stack<<<(unsigned)(n4 / 256), 256, 0, stream>>>(stack, new_elt, aw, out_stack);
}
